// UAPN_77567109366449
// MI455X (gfx1250) — compile-verified, matrix operand order checked by reading, not run on hardware
//
#include <hip/hip_runtime.h>
#include <hip/hip_bf16.h>
#include <math.h>

#define HDIM 256
#define WDIM 256
#define HW   65536

typedef __attribute__((ext_vector_type(16))) _Float16 v16h;
typedef __attribute__((ext_vector_type(8)))  _Float16 v8h;
typedef __attribute__((ext_vector_type(8)))  float    v8f;

// Inverse of the CDNA5 16-bit A/B fragment K mapping (ISA 7.12.2):
//   lane half = (k>>3)&1,  element j = (k&7) + 8*(k>>4)
// An aligned group of 8 consecutive k maps to ONE lane and 8 consecutive
// elements -> 16 contiguous LDS bytes (one b128).
__device__ __forceinline__ int frag_half(int k) { return (k >> 3) & 1; }
__device__ __forceinline__ int frag_elem(int k) { return (k & 7) + 8 * (k >> 4); }

__device__ __forceinline__ float act_apply(float v, int act) {
    switch (act) {
        case 1: return v > 0.f ? v : 0.f;                // relu
        case 2: return 1.f / (1.f + expf(-v));           // sigmoid
        case 3: return v > 0.f ? v : 0.2f * v;           // lrelu(0.2)
        default: return v;
    }
}

// =====================================================================
// Implicit-GEMM 3x3 conv, tap-major K (K = tap*Cin + c), optional
// per-pixel (spec, spat) modulation fused into the B-tile build.
// Weights arrive pre-swizzled in per-chunk fragment order, so the A
// stage is a contiguous b128 global->LDS copy. B stage: 8 channel loads
// -> one ds_store_b128. Fragment loads: one 32B LDS read per operand.
// =====================================================================
template <int MT, int NT>
__global__ __launch_bounds__(MT* NT * 32) void conv3x3_wmma(
    const _Float16* __restrict__ Wf,   // [Ktot/32][MT][32][16] fragment order
    const float* __restrict__ bias,    // [MT*16] or null
    const float* __restrict__ inA,     // [CinA][HW] f32
    const float* __restrict__ inB,     // [Cin-CinA][HW] f32 or null
    int CinA, int Cin, int lg2Cin,
    const float* __restrict__ spec,    // [Cin][HW] or null
    const float* __restrict__ spat,    // [9][HW]  or null
    float* __restrict__ out,           // [MT*16][HW]
    int act)
{
    __shared__ __align__(32) _Float16 sAf[MT][32][16];  // [mtile][lane][elem]
    __shared__ __align__(32) _Float16 sBf[NT][32][16];  // [ntile][lane][elem]

    constexpr int nthr = MT * NT * 32;
    constexpr int NPIX = NT * 16;
    const int tid = threadIdx.x;
    const int lane = tid & 31;
    const int waveId = tid >> 5;
    const int mtile = waveId / NT;
    const int ntile = waveId % NT;
    const int pixBase = blockIdx.x * NPIX;      // 64 | 256 -> single row
    const int hrow = pixBase >> 8;
    const int wbase = pixBase & 255;
    const int Ktot = Cin * 9;

    v8f acc = {};

    for (int K0 = 0; K0 < Ktot; K0 += 32) {
        const int tap = K0 >> lg2Cin;           // uniform per chunk (32 | Cin)
        const int dy = tap / 3 - 1;
        const int dx = tap % 3 - 1;
        const int hh = hrow + dy;
        const bool rowok = (hh >= 0) && (hh < HDIM);
        const int rowoff = hh << 8;

        __syncthreads();
        // ---- stage A: contiguous fragment-order copy, b128 granules ----
        const _Float16* Wc = Wf + (K0 >> 5) * (MT * 512);
        _Float16* sAflat = &sAf[0][0][0];
        for (int idx = tid; idx < MT * 64; idx += nthr)
            *(v8h*)(sAflat + idx * 8) = *(const v8h*)(Wc + idx * 8);

        // ---- stage B: 8 channels -> one lane's 16B, one b128 store ----
        for (int idx = tid; idx < NPIX * 4; idx += nthr) {
            int px = idx >> 2;
            int kg = (idx & 3) * 8;              // aligned group of 8 k
            int p = pixBase + px;
            int ww = wbase + px + dx;
            v8h t = {};
            if (rowok && ww >= 0 && ww < WDIM) {
                int ch0 = (K0 + kg) & (Cin - 1); // 8-aligned; no CinA cross
                const float* sp = (ch0 < CinA) ? inA + ch0 * HW
                                               : inB + (ch0 - CinA) * HW;
                const int goff = rowoff + ww;
                if (spec) {
                    const float* sc = spec + ch0 * HW + p;
                    float sv = spat[tap * HW + p];
#pragma unroll
                    for (int u = 0; u < 8; ++u)
                        t[u] = (_Float16)(sp[u * HW + goff] * sc[u * HW] * sv);
                } else {
#pragma unroll
                    for (int u = 0; u < 8; ++u)
                        t[u] = (_Float16)sp[u * HW + goff];
                }
            }
            *(v8h*)&sBf[px >> 4][frag_half(kg) * 16 + (px & 15)][frag_elem(kg)] = t;
        }
        __syncthreads();

        v16h a = *(const v16h*)&sAf[mtile][lane][0];
        v16h b = *(const v16h*)&sBf[ntile][lane][0];
        acc = __builtin_amdgcn_wmma_f32_16x16x32_f16(
            false, a, false, b, (short)0, acc, false, false);
    }

    const int half_id = lane >> 4;
    const int p = pixBase + ntile * 16 + (lane & 15);
#pragma unroll
    for (int j = 0; j < 8; ++j) {
        int m = mtile * 16 + half_id * 8 + j;
        float v = acc[j];
        if (bias) v += bias[m];
        out[m * HW + p] = act_apply(v, act);
    }
}

// =====================================================================
// 1x1 conv as GEMM over pixels; same staging scheme.
// In-place (out==in) is safe: all global reads precede all stores.
// =====================================================================
template <int MT, int NT>
__global__ __launch_bounds__(MT* NT * 32) void gemm1x1_wmma(
    const _Float16* __restrict__ Wf,   // [Cin/32][MT][32][16] fragment order
    const float* __restrict__ bias,    // [MT*16] or null
    const float* __restrict__ in,      // [Cin][HW]
    int Cin,
    float* __restrict__ out,           // [MT*16][HW]
    int act)
{
    __shared__ __align__(32) _Float16 sAf[MT][32][16];
    __shared__ __align__(32) _Float16 sBf[NT][32][16];

    constexpr int nthr = MT * NT * 32;
    constexpr int NPIX = NT * 16;
    const int tid = threadIdx.x;
    const int lane = tid & 31;
    const int waveId = tid >> 5;
    const int mtile = waveId / NT;
    const int ntile = waveId % NT;
    const int pixBase = blockIdx.x * NPIX;

    v8f acc = {};

    for (int K0 = 0; K0 < Cin; K0 += 32) {
        __syncthreads();
        const _Float16* Wc = Wf + (K0 >> 5) * (MT * 512);
        _Float16* sAflat = &sAf[0][0][0];
        for (int idx = tid; idx < MT * 64; idx += nthr)
            *(v8h*)(sAflat + idx * 8) = *(const v8h*)(Wc + idx * 8);

        for (int idx = tid; idx < NPIX * 4; idx += nthr) {
            int px = idx >> 2;
            int kg = (idx & 3) * 8;
            int p = pixBase + px;
            const float* ip = in + (K0 + kg) * HW + p;
            v8h t;
#pragma unroll
            for (int u = 0; u < 8; ++u) t[u] = (_Float16)ip[u * HW];
            *(v8h*)&sBf[px >> 4][frag_half(kg) * 16 + (px & 15)][frag_elem(kg)] = t;
            if (K0 + 32 < Cin)  // prefetch next K chunk (global_prefetch_b8)
                __builtin_prefetch(&in[(K0 + 32 + kg) * HW + p], 0, 0);
        }
        __syncthreads();

        v16h a = *(const v16h*)&sAf[mtile][lane][0];
        v16h b = *(const v16h*)&sBf[ntile][lane][0];
        acc = __builtin_amdgcn_wmma_f32_16x16x32_f16(
            false, a, false, b, (short)0, acc, false, false);
    }

    const int half_id = lane >> 4;
    const int p = pixBase + ntile * 16 + (lane & 15);
#pragma unroll
    for (int j = 0; j < 8; ++j) {
        int m = mtile * 16 + half_id * 8 + j;
        float v = acc[j];
        if (bias) v += bias[m];
        out[m * HW + p] = act_apply(v, act);
    }
}

// =====================================================================
// Weight swizzle kernels: f32 -> f16 in per-chunk fragment order
//   dst[((K/32)*MT + m/16)*32 + lane]*16 + j, K tap-major for 3x3.
// =====================================================================
__device__ __forceinline__ int swz_index(int m, int K, int MT) {
    int kl = K & 31;
    int lane = ((kl >> 3) & 1) * 16 + (m & 15);
    int j = (kl & 7) + 8 * (kl >> 4);
    return (((K >> 5) * MT + (m >> 4)) * 32 + lane) * 16 + j;
}

__global__ __launch_bounds__(256) void swz_w3x3_kernel(
    const float* __restrict__ src, _Float16* __restrict__ dst, int M, int Cin)
{
    int i = blockIdx.x * 256 + threadIdx.x;
    if (i >= M * Cin * 9) return;
    int m = i / (Cin * 9);
    int rem = i - m * Cin * 9;
    int c = rem / 9;
    int tap = rem - c * 9;
    dst[swz_index(m, tap * Cin + c, M >> 4)] = (_Float16)src[i];
}

__global__ __launch_bounds__(256) void swz_w1x1_kernel(
    const float* __restrict__ src, _Float16* __restrict__ dst, int M, int Cin)
{
    int i = blockIdx.x * 256 + threadIdx.x;
    if (i >= M * Cin) return;
    int m = i / Cin;
    int c = i - m * Cin;
    dst[swz_index(m, c, M >> 4)] = (_Float16)src[i];
}

// =====================================================================
// Small VALU kernels
// =====================================================================
__global__ __launch_bounds__(256) void ensemble_kernel(
    const float* __restrict__ x,        // [64][HW]
    const float* __restrict__ masks_u,  // [10][64]
    const float* __restrict__ w4,       // [4][64]
    const float* __restrict__ ob,       // [4]
    const float* __restrict__ ms,       // [4][HW]
    float* __restrict__ EU, float* __restrict__ mean)
{
    __shared__ float Wt[10][4][64];
    const int tid = threadIdx.x;
    for (int idx = tid; idx < 10 * 4 * 64; idx += 256) {
        int t = idx >> 8;
        int o = (idx >> 6) & 3;
        int c = idx & 63;
        float msk = (masks_u[t * 64 + c] < 0.8f) ? 1.f : 0.f;
        Wt[t][o][c] = msk * w4[o * 64 + c];
    }
    __syncthreads();

    const int p = blockIdx.x * 256 + tid;
    float xv[64];
#pragma unroll
    for (int c = 0; c < 64; ++c) xv[c] = x[c * HW + p];
    float msv[4], s1[4] = {0, 0, 0, 0}, s2[4] = {0, 0, 0, 0};
#pragma unroll
    for (int o = 0; o < 4; ++o) msv[o] = ms[o * HW + p];

    for (int t = 0; t < 10; ++t) {
        float a0 = ob[0], a1 = ob[1], a2 = ob[2], a3 = ob[3];
        for (int c = 0; c < 64; ++c) {
            float xc = xv[c];
            a0 += Wt[t][0][c] * xc;
            a1 += Wt[t][1][c] * xc;
            a2 += Wt[t][2][c] * xc;
            a3 += Wt[t][3][c] * xc;
        }
        float v0 = tanhf(a0) + msv[0];
        float v1 = tanhf(a1) + msv[1];
        float v2 = tanhf(a2) + msv[2];
        float v3 = tanhf(a3) + msv[3];
        s1[0] += v0; s2[0] += v0 * v0;
        s1[1] += v1; s2[1] += v1 * v1;
        s1[2] += v2; s2[2] += v2 * v2;
        s1[3] += v3; s2[3] += v3 * v3;
    }
#pragma unroll
    for (int o = 0; o < 4; ++o) {
        mean[o * HW + p] = s1[o] * 0.1f;
        EU[o * HW + p] = (s2[o] - s1[o] * s1[o] * 0.1f) * (1.f / 9.f);
    }
}

__global__ __launch_bounds__(256) void aue_kernel(
    const float* __restrict__ x,   // [64][HW]
    const float* __restrict__ w,   // [4][64][9]
    const float* __restrict__ b,   // [4]
    float* __restrict__ AU)
{
    __shared__ float sw[4 * 64 * 9];
    for (int idx = threadIdx.x; idx < 4 * 64 * 9; idx += 256) sw[idx] = w[idx];
    __syncthreads();

    const int p = blockIdx.x * 256 + threadIdx.x;
    const int h = p >> 8, wc = p & 255;
    float acc[4] = {b[0], b[1], b[2], b[3]};
    for (int c = 0; c < 64; ++c) {
#pragma unroll
        for (int tap = 0; tap < 9; ++tap) {
            int hh = h + tap / 3 - 1;
            int ww = wc + (tap % 3) - 1;
            float xv = 0.f;
            if (hh >= 0 && hh < HDIM && ww >= 0 && ww < WDIM)
                xv = x[c * HW + (hh << 8) + ww];
#pragma unroll
            for (int o = 0; o < 4; ++o) acc[o] += sw[(o * 64 + c) * 9 + tap] * xv;
        }
    }
#pragma unroll
    for (int o = 0; o < 4; ++o)
        AU[o * HW + p] = 1.f / (1.f + expf(-acc[o]));
}

__global__ __launch_bounds__(256) void spat_chain_kernel(
    const float* __restrict__ EU, int flip,
    const float* __restrict__ w0, const float* __restrict__ b0,  // (9,4,3,3),(9)
    const float* __restrict__ w1, const float* __restrict__ b1,  // (9,9)
    const float* __restrict__ w2, const float* __restrict__ b2,  // (9,9)
    float* __restrict__ spat)                                    // [9][HW]
{
    __shared__ float sw0[324], sw1[81], sw2[81], sb0[9], sb1[9], sb2[9];
    for (int i = threadIdx.x; i < 324; i += 256) sw0[i] = w0[i];
    for (int i = threadIdx.x; i < 81; i += 256) { sw1[i] = w1[i]; sw2[i] = w2[i]; }
    if (threadIdx.x < 9) {
        sb0[threadIdx.x] = b0[threadIdx.x];
        sb1[threadIdx.x] = b1[threadIdx.x];
        sb2[threadIdx.x] = b2[threadIdx.x];
    }
    __syncthreads();

    const int p = blockIdx.x * 256 + threadIdx.x;
    const int h = p >> 8, wc = p & 255;
    float a0[9];
#pragma unroll
    for (int o = 0; o < 9; ++o) a0[o] = sb0[o];
    for (int c = 0; c < 4; ++c) {
#pragma unroll
        for (int tap = 0; tap < 9; ++tap) {
            int hh = h + tap / 3 - 1;
            int ww = wc + (tap % 3) - 1;
            float u = 0.f;
            if (hh >= 0 && hh < HDIM && ww >= 0 && ww < WDIM) {
                u = EU[c * HW + (hh << 8) + ww];
                if (flip) u = 1.f - u;
            }
#pragma unroll
            for (int o = 0; o < 9; ++o) a0[o] += sw0[(o * 4 + c) * 9 + tap] * u;
        }
    }
#pragma unroll
    for (int o = 0; o < 9; ++o) a0[o] = a0[o] > 0.f ? a0[o] : 0.f;
    float a1[9];
#pragma unroll
    for (int o = 0; o < 9; ++o) {
        float v = sb1[o];
#pragma unroll
        for (int i = 0; i < 9; ++i) v += sw1[o * 9 + i] * a0[i];
        a1[o] = v > 0.f ? v : 0.f;
    }
#pragma unroll
    for (int o = 0; o < 9; ++o) {
        float v = sb2[o];
#pragma unroll
        for (int i = 0; i < 9; ++i) v += sw2[o * 9 + i] * a1[i];
        spat[o * HW + p] = 1.f / (1.f + expf(-v));
    }
}

__global__ __launch_bounds__(256) void spe0_kernel(
    const float* __restrict__ EU, int flip,
    const float* __restrict__ w0, const float* __restrict__ b0,  // (64,4),(64)
    float* __restrict__ s0out)                                   // [64][HW]
{
    __shared__ float sw[256], sb[64];
    for (int i = threadIdx.x; i < 256; i += 256) sw[i] = w0[i];
    if (threadIdx.x < 64) sb[threadIdx.x] = b0[threadIdx.x];
    __syncthreads();

    const int p = blockIdx.x * 256 + threadIdx.x;
    float u[4];
#pragma unroll
    for (int c = 0; c < 4; ++c) {
        float v = EU[c * HW + p];
        u[c] = flip ? (1.f - v) : v;
    }
    for (int m = 0; m < 64; ++m) {
        float v = sb[m];
#pragma unroll
        for (int c = 0; c < 4; ++c) v += sw[m * 4 + c] * u[c];
        s0out[m * HW + p] = v > 0.f ? v : 0.f;
    }
}

// =====================================================================
extern "C" void kernel_launch(void* const* d_in, const int* in_sizes, int n_in,
                              void* d_out, int out_size, void* d_ws, size_t ws_size,
                              hipStream_t stream)
{
    const float* F_1       = (const float*)d_in[0];
    const float* ms        = (const float*)d_in[1];
    const float* pan       = (const float*)d_in[2];
    const float* masks_u   = (const float*)d_in[3];
    const float* ue_conv_w = (const float*)d_in[4];
    const float* ue_conv_b = (const float*)d_in[5];
    const float* ue_out_w  = (const float*)d_in[6];
    const float* ue_out_b  = (const float*)d_in[7];
    const float* ue_aue_w  = (const float*)d_in[8];
    const float* ue_aue_b  = (const float*)d_in[9];
    const float* conv1_w   = (const float*)d_in[10];
    const float* conv1_b   = (const float*)d_in[11];
    const float* conv2_w   = (const float*)d_in[12];
    const float* conv2_b   = (const float*)d_in[13];
    const float* spa_w0[2] = {(const float*)d_in[14], (const float*)d_in[27]};
    const float* spa_b0[2] = {(const float*)d_in[15], (const float*)d_in[28]};
    const float* spa_w1[2] = {(const float*)d_in[16], (const float*)d_in[29]};
    const float* spa_b1[2] = {(const float*)d_in[17], (const float*)d_in[30]};
    const float* spa_w2[2] = {(const float*)d_in[18], (const float*)d_in[31]};
    const float* spa_b2[2] = {(const float*)d_in[19], (const float*)d_in[32]};
    const float* spe_w0[2] = {(const float*)d_in[20], (const float*)d_in[33]};
    const float* spe_b0[2] = {(const float*)d_in[21], (const float*)d_in[34]};
    const float* spe_w1[2] = {(const float*)d_in[22], (const float*)d_in[35]};
    const float* spe_b1[2] = {(const float*)d_in[23], (const float*)d_in[36]};
    const float* spe_w2[2] = {(const float*)d_in[24], (const float*)d_in[37]};
    const float* spe_b2[2] = {(const float*)d_in[25], (const float*)d_in[38]};
    const float* ua_w[2]   = {(const float*)d_in[26], (const float*)d_in[39]};

    float* outAU   = (float*)d_out;                 // 4*HW
    float* outEU   = outAU + 4 * HW;                // 4*HW
    float* outMean = outAU + 8 * HW;                // 4*HW
    float* outF    = outAU + 12 * HW;               // 32*HW

    // ---- workspace arena (floats), with reuse ----
    float* ws   = (float*)d_ws;
    float* x    = ws;                 // [64*HW]; reused later as y / F_2
    float* spat = ws + 64 * HW;       // [9*HW]
    float* spec = spat + 9 * HW;      // [64*HW] (s1 then spec, in place)
    float* tmp  = spec + 64 * HW;     // [64*HW] (s0)
    _Float16* wh = (_Float16*)(tmp + 64 * HW);
    _Float16* wh_ue  = wh;                             // 64*288
    _Float16* wh_ua[2]; wh_ua[0] = wh_ue + 64 * 288;   // 32*576 each
    wh_ua[1] = wh_ua[0] + 32 * 576;
    _Float16* wh_s1[2]; wh_s1[0] = wh_ua[1] + 32 * 576;
    _Float16* wh_s2[2]; wh_s2[0] = wh_s1[0] + 4096;
    wh_s1[1] = wh_s2[0] + 4096;
    wh_s2[1] = wh_s1[1] + 4096;
    _Float16* wh_c1 = wh_s2[1] + 4096;                 // 32*32
    _Float16* wh_c2 = wh_c1 + 1024;
    float* y  = x;                    // [32*HW]  (x dead after AU)
    float* F2 = x + 32 * HW;          // [32*HW]

    // 3x3 weights -> fragment-swizzled f16 (tap-major K)
    swz_w3x3_kernel<<<(64 * 288 + 255) / 256, 256, 0, stream>>>(
        ue_conv_w, wh_ue, 64, 32);
    swz_w3x3_kernel<<<(32 * 576 + 255) / 256, 256, 0, stream>>>(
        ua_w[0], wh_ua[0], 32, 64);
    swz_w3x3_kernel<<<(32 * 576 + 255) / 256, 256, 0, stream>>>(
        ua_w[1], wh_ua[1], 32, 64);
    // 1x1 weights -> fragment-swizzled f16
    swz_w1x1_kernel<<<16, 256, 0, stream>>>(spe_w1[0], wh_s1[0], 64, 64);
    swz_w1x1_kernel<<<16, 256, 0, stream>>>(spe_w2[0], wh_s2[0], 64, 64);
    swz_w1x1_kernel<<<16, 256, 0, stream>>>(spe_w1[1], wh_s1[1], 64, 64);
    swz_w1x1_kernel<<<16, 256, 0, stream>>>(spe_w2[1], wh_s2[1], 64, 64);
    swz_w1x1_kernel<<<4, 256, 0, stream>>>(conv1_w, wh_c1, 32, 32);
    swz_w1x1_kernel<<<4, 256, 0, stream>>>(conv2_w, wh_c2, 32, 32);

    // x = conv3x3(F_1, ue_conv_w) + b   (32 -> 64 channels)
    conv3x3_wmma<4, 4><<<HW / 64, 512, 0, stream>>>(
        wh_ue, ue_conv_b, F_1, nullptr, 32, 32, 5, nullptr, nullptr, x, 0);

    // EU / mean from T=10 masked projections
    ensemble_kernel<<<HW / 256, 256, 0, stream>>>(
        x, masks_u, ue_out_w, ue_out_b, ms, outEU, outMean);

    // AU = sigmoid(conv3x3(x, ue_aue_w))
    aue_kernel<<<HW / 256, 256, 0, stream>>>(x, ue_aue_w, ue_aue_b, outAU);

    for (int i = 0; i < 2; ++i) {
        // spatial attention (9 ch) and spectral attention (64 ch)
        spat_chain_kernel<<<HW / 256, 256, 0, stream>>>(
            outEU, i, spa_w0[i], spa_b0[i], spa_w1[i], spa_b1[i],
            spa_w2[i], spa_b2[i], spat);
        spe0_kernel<<<HW / 256, 256, 0, stream>>>(
            outEU, i, spe_w0[i], spe_b0[i], tmp);
        gemm1x1_wmma<4, 4><<<HW / 64, 512, 0, stream>>>(
            wh_s1[i], spe_b1[i], tmp, 64, spec, 1);        // relu
        gemm1x1_wmma<4, 4><<<HW / 64, 512, 0, stream>>>(
            wh_s2[i], spe_b2[i], spec, 64, spec, 2);       // sigmoid, in place

        // modulated 3x3 "einsum" conv: 64 -> 32, K = 576
        const float* inB = (i == 0) ? pan : F2;
        conv3x3_wmma<2, 4><<<HW / 64, 256, 0, stream>>>(
            wh_ua[i], nullptr, F_1, inB, 32, 64, 6, spec, spat, y, 0);

        if (i == 0) {
            gemm1x1_wmma<2, 4><<<HW / 64, 256, 0, stream>>>(
                wh_c1, conv1_b, y, 32, F2, 3);             // lrelu -> F_2
        } else {
            gemm1x1_wmma<2, 4><<<HW / 64, 256, 0, stream>>>(
                wh_c2, conv2_b, y, 32, outF, 3);           // lrelu -> F_out
        }
    }
}
